// cls_model_80350248174190
// MI455X (gfx1250) — matrix operand order repaired and checked by reading, not compiled, not run on hardware
//
#include <hip/hip_runtime.h>

typedef __attribute__((ext_vector_type(16))) _Float16 v16h;
typedef __attribute__((ext_vector_type(8)))  float    v8f;

#define BN_EPS 1e-5f

// ---- order-preserving float<->uint encoding (for max via integer atomics) ----
__device__ __forceinline__ unsigned enc_f32(float f) {
  unsigned u = __float_as_uint(f);
  return (u & 0x80000000u) ? ~u : (u | 0x80000000u);
}
__device__ __forceinline__ float dec_f32(unsigned u) {
  u = (u & 0x80000000u) ? (u & 0x7fffffffu) : ~u;
  return __uint_as_float(u);
}

// ---- WMMA fragment helpers (wave32, 16x16x32 f16 -> f32) ----
// A (16x32, 16-bit): lane l holds row m=l&15; lanes 16-31 use K base +16.
// elements 0..15 -> K = kb+0..15
__device__ __forceinline__ v16h load_a_frag(const _Float16* tile, int ld, int kstart, int lane) {
  const _Float16* p = tile + (size_t)(lane & 15) * ld + kstart + ((lane >> 4) << 4);
  v16h a;
#pragma unroll
  for (int e = 0; e < 8; ++e) a[e] = p[e];
#pragma unroll
  for (int e = 8; e < 16; ++e) a[e] = p[e];
  return a;
}
// B (32x16, 16-bit): lane l holds column n=l&15; lanes 16-31 use K base +16.
// B[k][n] sourced from row-major matrix rows (k contiguous within a row).
__device__ __forceinline__ v16h load_b_frag(const _Float16* base, int ldn, int kstart, int lane) {
  const _Float16* p = base + (size_t)(lane & 15) * ldn + kstart + ((lane >> 4) << 4);
  v16h b;
#pragma unroll
  for (int e = 0; e < 16; ++e) b[e] = p[e];
  return b;
}
__device__ __forceinline__ v8f wmma_f16(v16h a, v16h b, v8f c) {
  return __builtin_amdgcn_wmma_f32_16x16x32_f16(false, a, false, b, (short)0, c, false, false);
}

// ---------------- prep kernels ----------------
__global__ void __launch_bounds__(256)
prep_points_kernel(const float* __restrict__ pts, _Float16* __restrict__ xh0, int total) {
  int t = blockIdx.x * blockDim.x + threadIdx.x;
  if (t >= total * 32) return;
  int n = t >> 5, c = t & 31;
  xh0[t] = (c < 3) ? (_Float16)pts[n * 3 + c] : (_Float16)0.f;
}

__global__ void __launch_bounds__(256)
sqn_kernel(const _Float16* __restrict__ feat, int fstride, int fcoff, int C,
           float* __restrict__ sqn, int total) {
  int t = blockIdx.x * blockDim.x + threadIdx.x;
  if (t >= total) return;
  const _Float16* p = feat + (size_t)t * fstride + fcoff;
  float s = 0.f;
  for (int c = 0; c < C; ++c) { float x = (float)p[c]; s += x * x; }
  sqn[t] = s;
}

__global__ void __launch_bounds__(256)
wcvt_kernel(const float* __restrict__ W, _Float16* __restrict__ Wh, int O, int I, int KPAD) {
  int t = blockIdx.x * blockDim.x + threadIdx.x;
  if (t >= O * KPAD) return;
  int o = t / KPAD, c = t % KPAD;
  Wh[t] = (c < I) ? (_Float16)W[(size_t)o * I + c] : (_Float16)0.f;
}

__global__ void __launch_bounds__(256)
init_ymax_kernel(unsigned* __restrict__ ymax) {
  int t = blockIdx.x * blockDim.x + threadIdx.x;
  if (t < 8192) ymax[t] = 0u;  // encodes the minimum ordered value
}

// ---------------- KNN via WMMA Gram matrix ----------------
// score(i,j) = 2<xi,xj> - ||xj||^2  (row term dropped: constant per row)
template <int CPAD>
__global__ void __launch_bounds__(128)
knn_kernel(const _Float16* __restrict__ feat, int fstride, int fcoff,
           const float* __restrict__ sqn, int* __restrict__ knn_idx, int N) {
  const int b = blockIdx.y;
  const int n0 = blockIdx.x * 16;
  const int tid = threadIdx.x;
  const int lane = tid & 31, wave = tid >> 5;

  __shared__ _Float16 a_sh[16][CPAD];
  __shared__ float    sc_sh[16][128];
  __shared__ float    tv_sh[16][20];
  __shared__ int      ti_sh[16][20];

  const _Float16* fb = feat + (size_t)b * N * fstride + fcoff;
  for (int t = tid; t < 16 * CPAD; t += 128) {
    int m = t / CPAD, c = t % CPAD;
    a_sh[m][c] = fb[(size_t)(n0 + m) * fstride + c];
  }
  if (tid < 16) {
    for (int j = 0; j < 20; ++j) { tv_sh[tid][j] = -3.0e38f; ti_sh[tid][j] = 0; }
  }
  __syncthreads();

  v16h afrag[CPAD / 32];
#pragma unroll
  for (int kc = 0; kc < CPAD / 32; ++kc)
    afrag[kc] = load_a_frag(&a_sh[0][0], CPAD, kc * 32, lane);

  const float* sq = sqn + (size_t)b * N;
  for (int c0 = 0; c0 < N; c0 += 128) {
#pragma unroll
    for (int t = 0; t < 2; ++t) {
      const int sub = wave * 2 + t;
      const int col0 = c0 + sub * 16;
      v8f acc = {};
#pragma unroll
      for (int kc = 0; kc < CPAD / 32; ++kc) {
        v16h bf = load_b_frag(fb + (size_t)col0 * fstride, fstride, kc * 32, lane);
        acc = wmma_f16(afrag[kc], bf, acc);
      }
      const int n = lane & 15;
      const float sj = sq[col0 + n];
      const int mbase = (lane >> 4) << 3;
#pragma unroll
      for (int v = 0; v < 8; ++v)
        sc_sh[mbase + v][sub * 16 + n] = 2.0f * acc[v] - sj;
    }
    __syncthreads();
    if (tid < 16) {
      float* sv = tv_sh[tid];
      int*   si = ti_sh[tid];
      for (int c = 0; c < 128; ++c) {
        float val = sc_sh[tid][c];
        if (val > sv[19]) {
          int pos = 19;
          while (pos > 0 && sv[pos - 1] < val) {
            sv[pos] = sv[pos - 1]; si[pos] = si[pos - 1]; --pos;
          }
          sv[pos] = val; si[pos] = c0 + c;
        }
      }
    }
    __syncthreads();
  }
  for (int t = tid; t < 16 * 20; t += 128) {
    int m = t / 20, j = t % 20;
    knn_idx[((size_t)b * N + n0 + m) * 20 + j] = ti_sh[m][j];
  }
}

// ---------------- EdgeConv: gather edge features, WMMA GEMM, BN+LReLU, max over k ----------------
template <int C, int KPAD, int O>
__global__ void __launch_bounds__(256)
edgeconv_kernel(const _Float16* __restrict__ feat, int fstride, int fcoff,
                const int* __restrict__ knn_idx, const _Float16* __restrict__ Wh,
                const float* __restrict__ gp, const float* __restrict__ bp,
                const float* __restrict__ mp, const float* __restrict__ vp,
                _Float16* __restrict__ outh, int ostride, int ocoff, int N) {
  const int b = blockIdx.y;
  const int n0 = blockIdx.x * 4;
  const int tid = threadIdx.x;
  const int lane = tid & 31, wave = tid >> 5;

  __shared__ _Float16 f_sh[80][KPAD];          // 4 points x 20 neighbors
  __shared__ unsigned omax_sh[4][O];

  const _Float16* fb = feat + (size_t)b * N * fstride + fcoff;
  const int* ib = knn_idx + ((size_t)b * N + n0) * 20;

  for (int t = tid; t < 80 * KPAD; t += 256) {
    int r = t / KPAD, c = t % KPAD;
    int p = r / 20, j = r % 20;
    float val = 0.f;
    if (c < C) {
      int g = ib[p * 20 + j];
      val = (float)fb[(size_t)g * fstride + c] - (float)fb[(size_t)(n0 + p) * fstride + c];
    } else if (c < 2 * C) {
      val = (float)fb[(size_t)(n0 + p) * fstride + (c - C)];
    }
    f_sh[r][c] = (_Float16)val;
  }
  for (int t = tid; t < 4 * O; t += 256) omax_sh[t / O][t % O] = 0u;
  __syncthreads();

  const int NT = O / 16;                        // output-channel tiles
  for (int t = wave; t < 5 * NT; t += 8) {      // 5 M-tiles (80 rows)
    const int mi = t / NT, ni = t % NT;
    const int ocol = ni * 16 + (lane & 15);
    const float scale = gp[ocol] * rsqrtf(vp[ocol] + BN_EPS);
    const float shift = bp[ocol] - mp[ocol] * scale;
    v8f acc = {};
#pragma unroll
    for (int kc = 0; kc < KPAD / 32; ++kc) {
      v16h a   = load_a_frag(&f_sh[mi * 16][0], KPAD, kc * 32, lane);
      v16h bfr = load_b_frag(Wh + (size_t)(ni * 16) * KPAD, KPAD, kc * 32, lane);
      acc = wmma_f16(a, bfr, acc);
    }
    const int mbase = mi * 16 + ((lane >> 4) << 3);
#pragma unroll
    for (int v = 0; v < 8; ++v) {
      int r = mbase + v;
      float y = acc[v] * scale + shift;
      y = (y > 0.f) ? y : 0.2f * y;
      atomicMax(&omax_sh[r / 20][ocol], enc_f32(y));
    }
  }
  __syncthreads();
  _Float16* ob = outh + (size_t)b * N * ostride + ocoff;
  for (int t = tid; t < 4 * O; t += 256) {
    int p = t / O, o = t % O;
    ob[(size_t)(n0 + p) * ostride + o] = (_Float16)dec_f32(omax_sh[p][o]);
  }
}

// ---------------- Layer 5: (B*N,512)x(512,1024) + BN + LReLU + max over N ----------------
__global__ void __launch_bounds__(256)
layer5_kernel(const _Float16* __restrict__ h, const _Float16* __restrict__ W5h,
              const float* __restrict__ g5, const float* __restrict__ b5,
              const float* __restrict__ m5, const float* __restrict__ v5,
              unsigned* __restrict__ ymax, int N) {
  const int b = blockIdx.y;
  const int n0 = blockIdx.x * 32;
  const int tid = threadIdx.x, lane = tid & 31, wave = tid >> 5;

  __shared__ _Float16 h_sh[32][512];
  const unsigned* hu = (const unsigned*)(h + ((size_t)b * N + n0) * 512);
  unsigned* shu = (unsigned*)&h_sh[0][0];
  for (int t = tid; t < 32 * 512 / 2; t += 256) shu[t] = hu[t];
  __syncthreads();

  for (int t = wave; t < 2 * 64; t += 8) {
    const int mi = t >> 6, ni = t & 63;
    const int ocol = ni * 16 + (lane & 15);
    const float scale = g5[ocol] * rsqrtf(v5[ocol] + BN_EPS);
    const float shift = b5[ocol] - m5[ocol] * scale;
    v8f acc = {};
#pragma unroll
    for (int kc = 0; kc < 16; ++kc) {
      v16h a   = load_a_frag(&h_sh[mi * 16][0], 512, kc * 32, lane);
      v16h bfr = load_b_frag(W5h + (size_t)(ni * 16) * 512, 512, kc * 32, lane);
      acc = wmma_f16(a, bfr, acc);
    }
    float mx = -3.0e38f;
#pragma unroll
    for (int v = 0; v < 8; ++v) {
      float y = acc[v] * scale + shift;
      y = (y > 0.f) ? y : 0.2f * y;
      mx = fmaxf(mx, y);
    }
    atomicMax(&ymax[b * 1024 + ocol], enc_f32(mx));
  }
}

// ---------------- FC head (tiny) ----------------
__global__ void __launch_bounds__(256)
fc_head_kernel(const unsigned* __restrict__ ymax,
               const float* __restrict__ fcw1, const float* __restrict__ ag1,
               const float* __restrict__ ab1, const float* __restrict__ am1,
               const float* __restrict__ av1,
               const float* __restrict__ fcw2, const float* __restrict__ fcb2,
               const float* __restrict__ ag2, const float* __restrict__ ab2,
               const float* __restrict__ am2, const float* __restrict__ av2,
               const float* __restrict__ fcw3, const float* __restrict__ fcb3,
               float* __restrict__ out) {
  __shared__ float y_sh[8][1024];
  __shared__ float z1_sh[8][512];
  __shared__ float z2_sh[8][256];
  const int tid = threadIdx.x;
  for (int t = tid; t < 8 * 1024; t += 256) y_sh[t >> 10][t & 1023] = dec_f32(ymax[t]);
  __syncthreads();
  for (int t = tid; t < 8 * 512; t += 256) {
    int bi = t >> 9, o = t & 511;
    const float* w = fcw1 + (size_t)o * 1024;
    float s = 0.f;
    for (int c = 0; c < 1024; ++c) s += y_sh[bi][c] * w[c];
    float sc = ag1[o] * rsqrtf(av1[o] + BN_EPS);
    float yv = (s - am1[o]) * sc + ab1[o];
    z1_sh[bi][o] = yv > 0.f ? yv : 0.2f * yv;
  }
  __syncthreads();
  for (int t = tid; t < 8 * 256; t += 256) {
    int bi = t >> 8, o = t & 255;
    const float* w = fcw2 + (size_t)o * 512;
    float s = fcb2[o];
    for (int c = 0; c < 512; ++c) s += z1_sh[bi][c] * w[c];
    float sc = ag2[o] * rsqrtf(av2[o] + BN_EPS);
    float yv = (s - am2[o]) * sc + ab2[o];
    z2_sh[bi][o] = yv > 0.f ? yv : 0.2f * yv;
  }
  __syncthreads();
  for (int t = tid; t < 8 * 3; t += 256) {
    int bi = t / 3, o = t % 3;
    const float* w = fcw3 + (size_t)o * 256;
    float s = fcb3[o];
    for (int c = 0; c < 256; ++c) s += z2_sh[bi][c] * w[c];
    out[bi * 3 + o] = s;
  }
}

// ---------------- host launch ----------------
extern "C" void kernel_launch(void* const* d_in, const int* in_sizes, int n_in,
                              void* d_out, int out_size, void* d_ws, size_t ws_size,
                              hipStream_t stream) {
  (void)in_sizes; (void)n_in; (void)out_size; (void)ws_size;
  const int B = 8, N = 4096, BN = B * N;

  const float* pts = (const float*)d_in[0];
  const float* W1 = (const float*)d_in[1];
  const float *g1 = (const float*)d_in[2], *b1 = (const float*)d_in[3],
              *m1 = (const float*)d_in[4], *v1 = (const float*)d_in[5];
  const float* W2 = (const float*)d_in[6];
  const float *g2 = (const float*)d_in[7], *b2 = (const float*)d_in[8],
              *m2 = (const float*)d_in[9], *v2 = (const float*)d_in[10];
  const float* W3 = (const float*)d_in[11];
  const float *g3 = (const float*)d_in[12], *b3 = (const float*)d_in[13],
              *m3 = (const float*)d_in[14], *v3 = (const float*)d_in[15];
  const float* W4 = (const float*)d_in[16];
  const float *g4 = (const float*)d_in[17], *b4 = (const float*)d_in[18],
              *m4 = (const float*)d_in[19], *v4 = (const float*)d_in[20];
  const float* W5 = (const float*)d_in[21];
  const float *g5 = (const float*)d_in[22], *b5 = (const float*)d_in[23],
              *m5 = (const float*)d_in[24], *v5 = (const float*)d_in[25];
  const float* fcw1 = (const float*)d_in[26];
  const float *ag1 = (const float*)d_in[27], *ab1 = (const float*)d_in[28],
              *am1 = (const float*)d_in[29], *av1 = (const float*)d_in[30];
  const float* fcw2 = (const float*)d_in[31];
  const float* fcb2 = (const float*)d_in[32];
  const float *ag2 = (const float*)d_in[33], *ab2 = (const float*)d_in[34],
              *am2 = (const float*)d_in[35], *av2 = (const float*)d_in[36];
  const float* fcw3 = (const float*)d_in[37];
  const float* fcb3 = (const float*)d_in[38];

  // workspace carve-up (256B aligned)
  char* ws = (char*)d_ws;
  size_t off = 0;
  auto carve = [&](size_t bytes) -> void* {
    void* p = (void*)(ws + off);
    off += (bytes + 255) & ~(size_t)255;
    return p;
  };
  int*       idx  = (int*)carve((size_t)BN * 20 * sizeof(int));
  _Float16*  xh0  = (_Float16*)carve((size_t)BN * 32 * sizeof(_Float16));
  _Float16*  h    = (_Float16*)carve((size_t)BN * 512 * sizeof(_Float16));
  float*     sqn  = (float*)carve((size_t)BN * sizeof(float));
  _Float16*  w1h  = (_Float16*)carve((size_t)64 * 32 * sizeof(_Float16));
  _Float16*  w2h  = (_Float16*)carve((size_t)64 * 128 * sizeof(_Float16));
  _Float16*  w3h  = (_Float16*)carve((size_t)128 * 128 * sizeof(_Float16));
  _Float16*  w4h  = (_Float16*)carve((size_t)256 * 256 * sizeof(_Float16));
  _Float16*  w5h  = (_Float16*)carve((size_t)1024 * 512 * sizeof(_Float16));
  unsigned*  ymax = (unsigned*)carve((size_t)8192 * sizeof(unsigned));

  // prep
  prep_points_kernel<<<(BN * 32 + 255) / 256, 256, 0, stream>>>(pts, xh0, BN);
  wcvt_kernel<<<(64 * 32 + 255) / 256, 256, 0, stream>>>(W1, w1h, 64, 6, 32);
  wcvt_kernel<<<(64 * 128 + 255) / 256, 256, 0, stream>>>(W2, w2h, 64, 128, 128);
  wcvt_kernel<<<(128 * 128 + 255) / 256, 256, 0, stream>>>(W3, w3h, 128, 128, 128);
  wcvt_kernel<<<(256 * 256 + 255) / 256, 256, 0, stream>>>(W4, w4h, 256, 256, 256);
  wcvt_kernel<<<(1024 * 512 + 255) / 256, 256, 0, stream>>>(W5, w5h, 1024, 512, 512);
  init_ymax_kernel<<<32, 256, 0, stream>>>(ymax);

  dim3 kgrid(N / 16, B), egrid(N / 4, B);

  // EdgeConv 1 (C=3, from padded xh0) -> h[:, 0:64)
  sqn_kernel<<<(BN + 255) / 256, 256, 0, stream>>>(xh0, 32, 0, 3, sqn, BN);
  knn_kernel<32><<<kgrid, 128, 0, stream>>>(xh0, 32, 0, sqn, idx, N);
  edgeconv_kernel<3, 32, 64><<<egrid, 256, 0, stream>>>(
      xh0, 32, 0, idx, w1h, g1, b1, m1, v1, h, 512, 0, N);

  // EdgeConv 2 (C=64) -> h[:, 64:128)
  sqn_kernel<<<(BN + 255) / 256, 256, 0, stream>>>(h, 512, 0, 64, sqn, BN);
  knn_kernel<64><<<kgrid, 128, 0, stream>>>(h, 512, 0, sqn, idx, N);
  edgeconv_kernel<64, 128, 64><<<egrid, 256, 0, stream>>>(
      h, 512, 0, idx, w2h, g2, b2, m2, v2, h, 512, 64, N);

  // EdgeConv 3 (C=64) -> h[:, 128:256)
  sqn_kernel<<<(BN + 255) / 256, 256, 0, stream>>>(h, 512, 64, 64, sqn, BN);
  knn_kernel<64><<<kgrid, 128, 0, stream>>>(h, 512, 64, sqn, idx, N);
  edgeconv_kernel<64, 128, 128><<<egrid, 256, 0, stream>>>(
      h, 512, 64, idx, w3h, g3, b3, m3, v3, h, 512, 128, N);

  // EdgeConv 4 (C=128) -> h[:, 256:512)
  sqn_kernel<<<(BN + 255) / 256, 256, 0, stream>>>(h, 512, 128, 128, sqn, BN);
  knn_kernel<128><<<kgrid, 128, 0, stream>>>(h, 512, 128, sqn, idx, N);
  edgeconv_kernel<128, 256, 256><<<egrid, 256, 0, stream>>>(
      h, 512, 128, idx, w4h, g4, b4, m4, v4, h, 512, 256, N);

  // Layer 5 + global max pool over N
  layer5_kernel<<<dim3(N / 32, B), 256, 0, stream>>>(h, w5h, g5, b5, m5, v5, ymax, N);

  // FC head
  fc_head_kernel<<<1, 256, 0, stream>>>(ymax, fcw1, ag1, ab1, am1, av1,
                                        fcw2, fcb2, ag2, ab2, am2, av2,
                                        fcw3, fcb3, (float*)d_out);
}
